// RelMultiHeadAttentionLayer_85779086835883
// MI455X (gfx1250) — matrix operand order repaired and checked by reading, not compiled, not run on hardware
//
#include <hip/hip_runtime.h>
#include <hip/hip_bf16.h>

typedef unsigned short u16;
typedef __attribute__((ext_vector_type(16))) __bf16        bf16x16;
typedef __attribute__((ext_vector_type(8)))  float         f32x8;
typedef __attribute__((ext_vector_type(4)))  float         f32x4;
typedef __attribute__((ext_vector_type(4)))  unsigned int  u32x4;

#define HID   512
#define NH    8
#define HD    64
#define SEQ   64
#define BATCH 1024
#define ROWS  (BATCH * SEQ)   // 65536

// ---------- bf16 helpers ----------
__device__ __forceinline__ u16 f2bf(float f) {
  union { float f; unsigned u; } c; c.f = f;
  unsigned u = c.u;
  unsigned r = u + 0x7FFFu + ((u >> 16) & 1u);   // round-to-nearest-even
  return (u16)(r >> 16);
}
__device__ __forceinline__ unsigned pk2(float a, float b) {
  return (unsigned)f2bf(a) | ((unsigned)f2bf(b) << 16);
}

union FragU { u32x4 u[2]; bf16x16 v; };

// A fragment (16x32, M x K), row-major source with row stride rs (bf16 elems).
// ISA layout: lane<16 -> m=lane, K chunks {0..7},{16..23}; lane>=16 -> m=lane-16, {8..15},{24..31}
__device__ __forceinline__ bf16x16 load_frag_a(const u16* p, int rs, int lane) {
  int m  = lane & 15;
  int c0 = (lane < 16) ? 0 : 8;
  const u16* q = p + m * rs + c0;
  FragU f;
  f.u[0] = *(const u32x4*)(q);
  f.u[1] = *(const u32x4*)(q + 16);
  return f.v;
}
// B fragment (32x16, K x N) sourced from B^T stored [N][K] row-major, stride rs.
// ISA layout: lane n = lane&15, K chunks lane<16 -> {0..7},{16..23}; lane>=16 -> {8..15},{24..31}
__device__ __forceinline__ bf16x16 load_frag_b(const u16* p, int rs, int lane) {
  int n  = lane & 15;
  int c0 = (lane < 16) ? 0 : 8;
  const u16* q = p + n * rs + c0;
  FragU f;
  f.u[0] = *(const u32x4*)(q);
  f.u[1] = *(const u32x4*)(q + 16);
  return f.v;
}
__device__ __forceinline__ f32x8 wmma_bf16(bf16x16 a, bf16x16 b, f32x8 c) {
  return __builtin_amdgcn_wmma_f32_16x16x32_bf16(false, a, false, b, (short)0, c, false, false);
}

// ---------- prep: weights -> bf16, transposed to [N][K] ----------
__global__ __launch_bounds__(256) void prep_w_kernel(
    const float* __restrict__ Wq, const float* __restrict__ Wk,
    const float* __restrict__ Wv, const float* __restrict__ Wo,
    u16* __restrict__ WqT, u16* __restrict__ WkT,
    u16* __restrict__ WvT, u16* __restrict__ WoT) {
  const float* W; u16* WT;
  switch (blockIdx.y) {
    case 0:  W = Wq; WT = WqT; break;
    case 1:  W = Wk; WT = WkT; break;
    case 2:  W = Wv; WT = WvT; break;
    default: W = Wo; WT = WoT; break;
  }
  int idx = blockIdx.x * 256 + threadIdx.x;   // 0 .. 512*512-1
  int k = idx >> 9, n = idx & 511;
  WT[(size_t)n * 512 + k] = f2bf(W[(size_t)k * 512 + n]);
}

// ---------- prep: rel tables -> bf16, padded ----------
__global__ __launch_bounds__(256) void prep_rel_kernel(
    const float* __restrict__ rkx, const float* __restrict__ rky,
    const float* __restrict__ rvx, const float* __restrict__ rvy,
    u16* __restrict__ RKX, u16* __restrict__ RKY,
    u16* __restrict__ RVXT, u16* __restrict__ RVYT) {
  int t = threadIdx.x;
  // RKX/RKY: [16][32], rows 0..14 = rel_k table ([15][32]), row 15 zero
  for (int i = t; i < 16 * 32; i += 256) {
    int ix = i >> 5, d = i & 31;
    RKX[i] = (ix < 15) ? f2bf(rkx[ix * 32 + d]) : (u16)0;
    RKY[i] = (ix < 15) ? f2bf(rky[ix * 32 + d]) : (u16)0;
  }
  // RVXT/RVYT: [32][32] = rel_v^T padded: [d][ix], ix>=15 -> 0
  for (int i = t; i < 32 * 32; i += 256) {
    int d = i >> 5, ix = i & 31;
    RVXT[i] = (ix < 15) ? f2bf(rvx[ix * 32 + d]) : (u16)0;
    RVYT[i] = (ix < 15) ? f2bf(rvy[ix * 32 + d]) : (u16)0;
  }
}

// ---------- QKV projection GEMM: [65536,512] x [512,512] -> bf16 ----------
// Block tile 128M x 128N, 8 waves (4 M-strips x 2 N-strips), wave tile 32x64.
__global__ __launch_bounds__(256) void qkv_gemm_kernel(
    const float* __restrict__ inq, const float* __restrict__ ink, const float* __restrict__ inv,
    const u16* __restrict__ WqT, const u16* __restrict__ WkT, const u16* __restrict__ WvT,
    const float* __restrict__ bq, const float* __restrict__ bk, const float* __restrict__ bv,
    u16* __restrict__ qo, u16* __restrict__ ko, u16* __restrict__ vo) {
  __shared__ alignas(16) u16 At[128 * 32];
  __shared__ alignas(16) u16 Bt[128 * 32];

  const float* A; const u16* WT; const float* bias; u16* O;
  switch (blockIdx.z) {
    case 0:  A = inq; WT = WqT; bias = bq; O = qo; break;
    case 1:  A = ink; WT = WkT; bias = bk; O = ko; break;
    default: A = inv; WT = WvT; bias = bv; O = vo; break;
  }
  const int tid = threadIdx.x, lane = tid & 31, wv = tid >> 5;
  const int wm = wv & 3, wn = wv >> 2;               // 4x2 wave grid
  const int gm0 = blockIdx.x * 128, gn0 = blockIdx.y * 128;
  const int r = tid >> 1, hf = tid & 1;              // A staging: 128 rows x 2
  const int nb = tid >> 1, sg = tid & 1;             // B staging: 128 rows x 2

  f32x8 acc[2][4] = {};

  for (int kc = 0; kc < HID; kc += 32) {
    // stage A tile (f32 -> bf16)
    const float* ap = A + (size_t)(gm0 + r) * HID + kc + hf * 16;
    f32x4 f0 = *(const f32x4*)(ap + 0);
    f32x4 f1 = *(const f32x4*)(ap + 4);
    f32x4 f2 = *(const f32x4*)(ap + 8);
    f32x4 f3 = *(const f32x4*)(ap + 12);
    u32x4 p0, p1;
    p0.x = pk2(f0.x, f0.y); p0.y = pk2(f0.z, f0.w);
    p0.z = pk2(f1.x, f1.y); p0.w = pk2(f1.z, f1.w);
    p1.x = pk2(f2.x, f2.y); p1.y = pk2(f2.z, f2.w);
    p1.z = pk2(f3.x, f3.y); p1.w = pk2(f3.z, f3.w);
    *(u32x4*)&At[r * 32 + hf * 16 + 0] = p0;
    *(u32x4*)&At[r * 32 + hf * 16 + 8] = p1;
    // stage B tile (already bf16, [N][K] layout): 128 rows x 32 cols
    const u16* bp = WT + (size_t)(gn0 + nb) * HID + kc + sg * 16;
    *(u32x4*)&Bt[nb * 32 + sg * 16 + 0] = *(const u32x4*)(bp + 0);
    *(u32x4*)&Bt[nb * 32 + sg * 16 + 8] = *(const u32x4*)(bp + 8);
    __syncthreads();

    bf16x16 a0 = load_frag_a(&At[(wm * 32 + 0)  * 32], 32, lane);
    bf16x16 a1 = load_frag_a(&At[(wm * 32 + 16) * 32], 32, lane);
    bf16x16 b0 = load_frag_b(&Bt[(wn * 64 + 0)  * 32], 32, lane);
    bf16x16 b1 = load_frag_b(&Bt[(wn * 64 + 16) * 32], 32, lane);
    bf16x16 b2 = load_frag_b(&Bt[(wn * 64 + 32) * 32], 32, lane);
    bf16x16 b3 = load_frag_b(&Bt[(wn * 64 + 48) * 32], 32, lane);
    acc[0][0] = wmma_bf16(a0, b0, acc[0][0]);
    acc[0][1] = wmma_bf16(a0, b1, acc[0][1]);
    acc[0][2] = wmma_bf16(a0, b2, acc[0][2]);
    acc[0][3] = wmma_bf16(a0, b3, acc[0][3]);
    acc[1][0] = wmma_bf16(a1, b0, acc[1][0]);
    acc[1][1] = wmma_bf16(a1, b1, acc[1][1]);
    acc[1][2] = wmma_bf16(a1, b2, acc[1][2]);
    acc[1][3] = wmma_bf16(a1, b3, acc[1][3]);
    __syncthreads();
  }

  // store C (bf16) + bias
  for (int a = 0; a < 2; ++a)
    for (int bn = 0; bn < 4; ++bn) {
      int n = gn0 + wn * 64 + bn * 16 + (lane & 15);
      float bvv = bias[n];
      for (int i = 0; i < 8; ++i) {
        int m = gm0 + wm * 32 + a * 16 + i + ((lane >= 16) ? 8 : 0);
        O[(size_t)m * HID + n] = f2bf(acc[a][bn][i] + bvv);
      }
    }
}

// ---------- attention: one workgroup (4 waves) per (b,h) ----------
__global__ __launch_bounds__(128) void attn_kernel(
    const u16* __restrict__ qbf, const u16* __restrict__ kbf, const u16* __restrict__ vbf,
    const u16* __restrict__ RKX, const u16* __restrict__ RKY,
    const u16* __restrict__ RVXT, const u16* __restrict__ RVYT,
    u16* __restrict__ xbf) {
  __shared__ alignas(16) u16   Qs[64 * 64];    // rq; reused as P (bf16) after softmax
  __shared__ alignas(16) u16   Ks[64 * 64];
  __shared__ alignas(16) u16   VsT[64 * 64];   // [d][k] (rv transposed)
  __shared__ alignas(16) float Sf[64 * 64];    // scores / probabilities (f32)
  __shared__ float pxL[64 * 16];
  __shared__ float pyL[64 * 16];
  __shared__ alignas(16) u16   axL[64 * 32];   // K padded to 32
  __shared__ alignas(16) u16   ayL[64 * 32];
  __shared__ float redmax[64 * 2];
  __shared__ float redsum[64 * 2];

  const int tid = threadIdx.x, lane = tid & 31, w = tid >> 5;
  const int bidx = blockIdx.x >> 3, h = blockIdx.x & 7;
  const size_t base = (size_t)bidx * SEQ * HID + (size_t)h * HD;

  // ---- stage rq, rk, rv(T) ----
  {
    int rr = tid >> 1, hf2 = tid & 1;
    const u16* qp = qbf + base + (size_t)rr * HID + hf2 * 32;
    const u16* kp = kbf + base + (size_t)rr * HID + hf2 * 32;
    const u16* vp = vbf + base + (size_t)rr * HID + hf2 * 32;
    *(u32x4*)&Qs[rr * 64 + hf2 * 32 + 0]  = *(const u32x4*)(qp + 0);
    *(u32x4*)&Qs[rr * 64 + hf2 * 32 + 8]  = *(const u32x4*)(qp + 8);
    *(u32x4*)&Qs[rr * 64 + hf2 * 32 + 16] = *(const u32x4*)(qp + 16);
    *(u32x4*)&Qs[rr * 64 + hf2 * 32 + 24] = *(const u32x4*)(qp + 24);
    *(u32x4*)&Ks[rr * 64 + hf2 * 32 + 0]  = *(const u32x4*)(kp + 0);
    *(u32x4*)&Ks[rr * 64 + hf2 * 32 + 8]  = *(const u32x4*)(kp + 8);
    *(u32x4*)&Ks[rr * 64 + hf2 * 32 + 16] = *(const u32x4*)(kp + 16);
    *(u32x4*)&Ks[rr * 64 + hf2 * 32 + 24] = *(const u32x4*)(kp + 24);
    union { u32x4 u; u16 s[8]; } va, vb2, vc2, vd2;
    va.u  = *(const u32x4*)(vp + 0);
    vb2.u = *(const u32x4*)(vp + 8);
    vc2.u = *(const u32x4*)(vp + 16);
    vd2.u = *(const u32x4*)(vp + 24);
    for (int j = 0; j < 8; ++j) {
      VsT[(hf2 * 32 + j)      * 64 + rr] = va.s[j];
      VsT[(hf2 * 32 + 8 + j)  * 64 + rr] = vb2.s[j];
      VsT[(hf2 * 32 + 16 + j) * 64 + rr] = vc2.s[j];
      VsT[(hf2 * 32 + 24 + j) * 64 + rr] = vd2.s[j];
    }
  }
  __syncthreads();

  // ---- px = rq[:,0:32] @ rel_k_x^T,  py = rq[:,32:64] @ rel_k_y^T (wave w -> q strip w) ----
  {
    bf16x16 aqx = load_frag_a(&Qs[(w * 16) * 64 + 0],  64, lane);
    bf16x16 aqy = load_frag_a(&Qs[(w * 16) * 64 + 32], 64, lane);
    bf16x16 bx  = load_frag_b(RKX, 32, lane);
    bf16x16 by  = load_frag_b(RKY, 32, lane);
    f32x8 cx = {}; cx = wmma_bf16(aqx, bx, cx);
    f32x8 cy = {}; cy = wmma_bf16(aqy, by, cy);
    int nn = lane & 15, hi = (lane >= 16) ? 8 : 0;
    for (int i = 0; i < 8; ++i) {
      int m = w * 16 + i + hi;
      pxL[m * 16 + nn] = cx[i];
      pyL[m * 16 + nn] = cy[i];
    }
  }

  // ---- S1 = rq @ rk^T (wave w -> q strip w, all 4 k tiles) ----
  f32x8 acc[4] = {};
  {
    bf16x16 a0 = load_frag_a(&Qs[(w * 16) * 64 + 0],  64, lane);
    bf16x16 a1 = load_frag_a(&Qs[(w * 16) * 64 + 32], 64, lane);
    for (int nt = 0; nt < 4; ++nt) {
      bf16x16 b0 = load_frag_b(&Ks[(nt * 16) * 64 + 0],  64, lane);
      bf16x16 b1 = load_frag_b(&Ks[(nt * 16) * 64 + 32], 64, lane);
      acc[nt] = wmma_bf16(a0, b0, acc[nt]);
      acc[nt] = wmma_bf16(a1, b1, acc[nt]);
    }
  }
  __syncthreads();   // px/py visible across lanes

  // ---- scores = (S1 + px[dx] + py[dy]) / 8 ----
  {
    int nn = lane & 15, hi = (lane >= 16) ? 8 : 0;
    for (int nt = 0; nt < 4; ++nt) {
      int n = nt * 16 + nn;
      int kx = n >> 3, ky = n & 7;
      for (int i = 0; i < 8; ++i) {
        int m = w * 16 + i + hi;
        int qx = m >> 3, qy = m & 7;
        float v = acc[nt][i] + pxL[m * 16 + (kx - qx + 7)] + pyL[m * 16 + (ky - qy + 7)];
        Sf[m * 64 + n] = v * 0.125f;
      }
    }
  }
  __syncthreads();

  // ---- softmax over k (2 threads per row), then P->bf16 in Qs, and ax/ay bins ----
  {
    int rr = tid & 63, hf2 = tid >> 6;
    float mx = -1e30f;
    for (int j = 0; j < 32; ++j) mx = fmaxf(mx, Sf[rr * 64 + hf2 * 32 + j]);
    redmax[rr * 2 + hf2] = mx;
    __syncthreads();
    float M = fmaxf(redmax[rr * 2], redmax[rr * 2 + 1]);
    float sm = 0.f;
    for (int j = 0; j < 32; ++j) {
      float e = __expf(Sf[rr * 64 + hf2 * 32 + j] - M);
      Sf[rr * 64 + hf2 * 32 + j] = e;
      sm += e;
    }
    redsum[rr * 2 + hf2] = sm;
    __syncthreads();
    float inv = 1.f / (redsum[rr * 2] + redsum[rr * 2 + 1]);
    for (int j = 0; j < 32; ++j) {
      float p = Sf[rr * 64 + hf2 * 32 + j] * inv;
      Sf[rr * 64 + hf2 * 32 + j] = p;
      Qs[rr * 64 + hf2 * 32 + j] = f2bf(p);   // P (bf16), reusing Qs
    }
    __syncthreads();
    if (hf2 == 0) {          // ax[q,ix] = sum_ky P[q, (qx+ix-7)*8+ky]
      int qx = rr >> 3;
      for (int ix = 0; ix < 16; ++ix) {
        int kx = qx + ix - 7;
        float s = 0.f;
        if (kx >= 0 && kx < 8)
          for (int ky2 = 0; ky2 < 8; ++ky2) s += Sf[rr * 64 + kx * 8 + ky2];
        axL[rr * 32 + ix] = f2bf(s);
        axL[rr * 32 + 16 + ix] = 0;
      }
    } else {                 // ay[q,iy] = sum_kx P[q, kx*8 + qy+iy-7]
      int qy = rr & 7;
      for (int iy = 0; iy < 16; ++iy) {
        int ky = qy + iy - 7;
        float s = 0.f;
        if (ky >= 0 && ky < 8)
          for (int kx2 = 0; kx2 < 8; ++kx2) s += Sf[rr * 64 + kx2 * 8 + ky];
        ayL[rr * 32 + iy] = f2bf(s);
        ayL[rr * 32 + 16 + iy] = 0;
      }
    }
    __syncthreads();
  }

  // ---- O = P @ rv  +  ax @ rel_v_x  (+)  ay @ rel_v_y ----
  f32x8 oc[4] = {};
  {
    bf16x16 p0 = load_frag_a(&Qs[(w * 16) * 64 + 0],  64, lane);
    bf16x16 p1 = load_frag_a(&Qs[(w * 16) * 64 + 32], 64, lane);
    for (int nt = 0; nt < 4; ++nt) {
      bf16x16 b0 = load_frag_b(&VsT[(nt * 16) * 64 + 0],  64, lane);
      bf16x16 b1 = load_frag_b(&VsT[(nt * 16) * 64 + 32], 64, lane);
      oc[nt] = wmma_bf16(p0, b0, oc[nt]);
      oc[nt] = wmma_bf16(p1, b1, oc[nt]);
    }
    bf16x16 ax = load_frag_a(&axL[(w * 16) * 32], 32, lane);
    bf16x16 ay = load_frag_a(&ayL[(w * 16) * 32], 32, lane);
    for (int nt = 0; nt < 2; ++nt) {
      bf16x16 bx = load_frag_b(RVXT + nt * 16 * 32, 32, lane);
      oc[nt] = wmma_bf16(ax, bx, oc[nt]);                // d 0..31
      bf16x16 by = load_frag_b(RVYT + nt * 16 * 32, 32, lane);
      oc[nt + 2] = wmma_bf16(ay, by, oc[nt + 2]);        // d 32..63
    }
  }

  // ---- store x (bf16) at [b, s, h*64 + d] ----
  {
    int nn = lane & 15, hi = (lane >= 16) ? 8 : 0;
    for (int nt = 0; nt < 4; ++nt)
      for (int i = 0; i < 8; ++i) {
        int m = w * 16 + i + hi;
        int d = nt * 16 + nn;
        xbf[base + (size_t)m * HID + d] = f2bf(oc[nt][i]);
      }
  }
}

// ---------- output projection: x(bf16) @ Wo + bo -> f32 ----------
// Block tile 128M x 128N, wave tile 32x64 (same scheme as qkv_gemm).
__global__ __launch_bounds__(256) void oproj_gemm_kernel(
    const u16* __restrict__ X, const u16* __restrict__ WoT,
    const float* __restrict__ bo, float* __restrict__ out) {
  __shared__ alignas(16) u16 At[128 * 32];
  __shared__ alignas(16) u16 Bt[128 * 32];
  const int tid = threadIdx.x, lane = tid & 31, wv = tid >> 5;
  const int wm = wv & 3, wn = wv >> 2;
  const int gm0 = blockIdx.x * 128, gn0 = blockIdx.y * 128;
  const int r = tid >> 1, hf = tid & 1;

  f32x8 acc[2][4] = {};

  for (int kc = 0; kc < HID; kc += 32) {
    const u16* ap = X + (size_t)(gm0 + r) * HID + kc + hf * 16;
    *(u32x4*)&At[r * 32 + hf * 16 + 0] = *(const u32x4*)(ap + 0);
    *(u32x4*)&At[r * 32 + hf * 16 + 8] = *(const u32x4*)(ap + 8);
    const u16* bp = WoT + (size_t)(gn0 + r) * HID + kc + hf * 16;
    *(u32x4*)&Bt[r * 32 + hf * 16 + 0] = *(const u32x4*)(bp + 0);
    *(u32x4*)&Bt[r * 32 + hf * 16 + 8] = *(const u32x4*)(bp + 8);
    __syncthreads();

    bf16x16 a0 = load_frag_a(&At[(wm * 32 + 0)  * 32], 32, lane);
    bf16x16 a1 = load_frag_a(&At[(wm * 32 + 16) * 32], 32, lane);
    bf16x16 b0 = load_frag_b(&Bt[(wn * 64 + 0)  * 32], 32, lane);
    bf16x16 b1 = load_frag_b(&Bt[(wn * 64 + 16) * 32], 32, lane);
    bf16x16 b2 = load_frag_b(&Bt[(wn * 64 + 32) * 32], 32, lane);
    bf16x16 b3 = load_frag_b(&Bt[(wn * 64 + 48) * 32], 32, lane);
    acc[0][0] = wmma_bf16(a0, b0, acc[0][0]);
    acc[0][1] = wmma_bf16(a0, b1, acc[0][1]);
    acc[0][2] = wmma_bf16(a0, b2, acc[0][2]);
    acc[0][3] = wmma_bf16(a0, b3, acc[0][3]);
    acc[1][0] = wmma_bf16(a1, b0, acc[1][0]);
    acc[1][1] = wmma_bf16(a1, b1, acc[1][1]);
    acc[1][2] = wmma_bf16(a1, b2, acc[1][2]);
    acc[1][3] = wmma_bf16(a1, b3, acc[1][3]);
    __syncthreads();
  }

  for (int a = 0; a < 2; ++a)
    for (int bn = 0; bn < 4; ++bn) {
      int n = gn0 + wn * 64 + bn * 16 + (lane & 15);
      float bvv = bo[n];
      for (int i = 0; i < 8; ++i) {
        int m = gm0 + wm * 32 + a * 16 + i + ((lane >= 16) ? 8 : 0);
        out[(size_t)m * HID + n] = acc[a][bn][i] + bvv;
      }
    }
}

// ---------- host launch ----------
extern "C" void kernel_launch(void* const* d_in, const int* in_sizes, int n_in,
                              void* d_out, int out_size, void* d_ws, size_t ws_size,
                              hipStream_t stream) {
  (void)in_sizes; (void)n_in; (void)out_size; (void)ws_size;
  const float* query = (const float*)d_in[0];
  const float* key_  = (const float*)d_in[1];
  const float* value = (const float*)d_in[2];
  const float* Wq = (const float*)d_in[3];
  const float* bq = (const float*)d_in[4];
  const float* Wk = (const float*)d_in[5];
  const float* bk = (const float*)d_in[6];
  const float* Wv = (const float*)d_in[7];
  const float* bv = (const float*)d_in[8];
  const float* Wo = (const float*)d_in[9];
  const float* bo = (const float*)d_in[10];
  const float* rkx = (const float*)d_in[11];
  const float* rky = (const float*)d_in[12];
  const float* rvx = (const float*)d_in[13];
  const float* rvy = (const float*)d_in[14];
  float* out = (float*)d_out;

  char* ws = (char*)d_ws;
  u16* WqT  = (u16*)(ws + (size_t)0);
  u16* WkT  = (u16*)(ws + (size_t)512 * 1024);
  u16* WvT  = (u16*)(ws + (size_t)1024 * 1024);
  u16* WoT  = (u16*)(ws + (size_t)1536 * 1024);
  u16* RKX  = (u16*)(ws + (size_t)2048 * 1024);
  u16* RKY  = RKX + 16 * 32;
  u16* RVXT = RKY + 16 * 32;
  u16* RVYT = RVXT + 32 * 32;
  u16* qbf  = (u16*)(ws + (size_t)4 * 1024 * 1024);
  u16* kbf  = qbf + (size_t)ROWS * HID;
  u16* vbf  = kbf + (size_t)ROWS * HID;
  u16* xbf  = vbf + (size_t)ROWS * HID;

  prep_w_kernel<<<dim3(1024, 4), 256, 0, stream>>>(Wq, Wk, Wv, Wo, WqT, WkT, WvT, WoT);
  prep_rel_kernel<<<1, 256, 0, stream>>>(rkx, rky, rvx, rvy, RKX, RKY, RVXT, RVYT);
  qkv_gemm_kernel<<<dim3(ROWS / 128, HID / 128, 3), 256, 0, stream>>>(
      query, key_, value, WqT, WkT, WvT, bq, bk, bv, qbf, kbf, vbf);
  attn_kernel<<<dim3(BATCH * NH), 128, 0, stream>>>(
      qbf, kbf, vbf, RKX, RKY, RVXT, RVYT, xbf);
  oproj_gemm_kernel<<<dim3(ROWS / 128, HID / 128), 256, 0, stream>>>(xbf, WoT, bo, out);
}
